// AetherLinearPTQ_87144886436378
// MI455X (gfx1250) — matrix operand order repaired and checked by reading, not compiled, not run on hardware
//
#include <hip/hip_runtime.h>
#include <hip/hip_bf16.h>
#include <math.h>

typedef __bf16 v8bf  __attribute__((ext_vector_type(8)));
typedef __bf16 v16bf __attribute__((ext_vector_type(16)));
typedef float  v8f   __attribute__((ext_vector_type(8)));

// 16-byte int vector matching the async builtin's parameter type
typedef int v4i_vs __attribute__((vector_size(16)));

// ---------------------------------------------------------------------------
// CDNA5 async global->LDS copy (ASYNCcnt path). LDS pointers are cast to AS(3)
// ONCE per thread (outside the loop) so the flat->local null-check select
// doesn't appear in the hot loop. Fallback: synchronous generic copy.
// ---------------------------------------------------------------------------
#if __has_builtin(__builtin_amdgcn_global_load_async_to_lds_b128) && \
    __has_builtin(__builtin_amdgcn_s_wait_asynccnt)
typedef __attribute__((address_space(3))) v4i_vs* ldsv4_t;
#define TO_LDSV4(p) ((ldsv4_t)(p))
#define ASYNC_CP16(g, l)                                                     \
    __builtin_amdgcn_global_load_async_to_lds_b128(                          \
        (v4i_vs*)const_cast<unsigned short*>(g), (l), 0, 0)
#define WAIT_ASYNC(n) __builtin_amdgcn_s_wait_asynccnt(n)
#else
typedef uint4* ldsv4_t;
#define TO_LDSV4(p) ((uint4*)(p))
#define ASYNC_CP16(g, l) (*(l) = *(const uint4*)(g))
#define WAIT_ASYNC(n) ((void)0)
#endif

// ---------------------------------------------------------------------------
// AF16 quantization (faithful port of float_to_af_tensor, bits=16 path).
// All power-of-2 scalings via ldexpf / hex-float literals -> exact in fp32.
// rintf == round-half-even under default RN mode (matches torch.round).
// ---------------------------------------------------------------------------
__device__ __forceinline__ float af16_quant(float xf) {
    float val = fabsf(xf);
    int er = 0;
    (void)frexpf(val, &er);                         // val = m * 2^er, m in [0.5,1)
    float e_b2  = (val == 0.0f) ? -256.0f : (float)er - 1.0f;
    float e_true = floorf(e_b2 * 0.5f);             // base-4 exponent
    float E = e_true + 63.0f;
    bool  is_sub = (E <= 0.0f);
    float M_sub  = (val * 0x1p65f) * 0x1p65f;       // val * 2^130, split exact
    float e_safe = fmaxf(e_true, -64.0f);
    float M_norm = ldexpf(val, (int)(-2.0f * e_safe)) * 64.0f;  // val/4^e * 64
    float Mf = is_sub ? M_sub : M_norm;
    E = is_sub ? 0.0f : E;
    E = fminf(E, 127.0f);
    float Mi = rintf(Mf);                           // round half to even
    bool ovf = (Mi >= 256.0f) && (E > 0.0f);
    if (ovf) { Mi = floorf(Mi * 0.25f); E += 1.0f; }
    if ((E == 0.0f) && (Mi >= 64.0f)) E = 1.0f;     // subnormal rounded up
    float vsub  = (Mi * 0x1p-65f) * 0x1p-65f;       // Mi * 2^-130, split exact
    float vnorm = ldexpf(Mi * (1.0f / 64.0f), 2 * ((int)E - 63));
    float outv = (E == 0.0f) ? vsub : vnorm;
    if (E >= 127.0f) outv = __builtin_inff();
    outv = (xf < 0.0f) ? -outv : outv;
    if (xf == 0.0f) outv = 0.0f;
    return outv;
}

__device__ __forceinline__ unsigned short f32_to_bf16_rne(float f) {
    unsigned int u = __float_as_uint(f);
    unsigned int r = u + 0x7FFFu + ((u >> 16) & 1u);
    return (unsigned short)(r >> 16);
}

// Pass 1: x fp32 -> AF16 -> bf16 (exact: AF16 normals fit in bf16)
__global__ void __launch_bounds__(256)
quant_x_kernel(const float* __restrict__ x, unsigned short* __restrict__ xq,
               long long n) {
    long long i4 = ((long long)blockIdx.x * 256 + threadIdx.x) * 4;
    if (i4 + 3 < n) {
        float4 v = *(const float4*)(x + i4);
        ushort4 o;
        o.x = f32_to_bf16_rne(af16_quant(v.x));
        o.y = f32_to_bf16_rne(af16_quant(v.y));
        o.z = f32_to_bf16_rne(af16_quant(v.z));
        o.w = f32_to_bf16_rne(af16_quant(v.w));
        *(ushort4*)(xq + i4) = o;
    }
}

// Pass 2: weight fp32 -> bf16 (weight is already AF16-quantized -> exact)
__global__ void __launch_bounds__(256)
w_bf16_kernel(const float* __restrict__ w, unsigned short* __restrict__ wq,
              long long n) {
    long long i4 = ((long long)blockIdx.x * 256 + threadIdx.x) * 4;
    if (i4 + 3 < n) {
        float4 v = *(const float4*)(w + i4);
        ushort4 o;
        o.x = f32_to_bf16_rne(v.x);
        o.y = f32_to_bf16_rne(v.y);
        o.z = f32_to_bf16_rne(v.z);
        o.w = f32_to_bf16_rne(v.w);
        *(ushort4*)(wq + i4) = o;
    }
}

// ---------------------------------------------------------------------------
// Pass 3: out[m][n] = sum_k Xq[m][k] * Wq[n][k] + bias[n]   (bf16 WMMA, f32 acc)
// 256 thr = 8 waves; block tile 128x128; wave tile 32x64 (2x4 WMMA tiles).
// Ping-pong LDS buffers (compile-time indices) fed by async global->LDS b128.
// ---------------------------------------------------------------------------
#define BM  128
#define BN  128
#define BKK 32
#define LDT 40   // LDS row stride (bf16 elems): 80B rows, breaks bank conflicts

// One 32-K MMA step for this wave from the given LDS tile bases.
__device__ __forceinline__ void mma_step(const unsigned short* Ab,
                                         const unsigned short* Bb,
                                         int wm, int wn, int l15, int half,
                                         v8f (&acc)[2][4]) {
    // A frag (16x32 bf16): lane half picks K {0..7,16..23} or {8..15,24..31}
    v16bf afrag[2];
#pragma unroll
    for (int tm = 0; tm < 2; ++tm) {
        const unsigned short* ap = Ab + (wm + tm * 16 + l15) * LDT + half * 8;
        v8bf lo = *(const v8bf*)(ap);        // K = half*8 + 0..7
        v8bf hi = *(const v8bf*)(ap + 16);   // K = 16 + half*8 + 0..7
        afrag[tm] = __builtin_shufflevector(lo, hi,
            0, 1, 2, 3, 4, 5, 6, 7, 8, 9, 10, 11, 12, 13, 14, 15);
    }
    // B frag (32x16 bf16): N = l15, K = half*8 + 0..7, 16 + half*8 + 0..7
    v16bf bfrag[4];
#pragma unroll
    for (int tn = 0; tn < 4; ++tn) {
        const unsigned short* bp = Bb + (wn + tn * 16 + l15) * LDT + half * 8;
        v8bf lo = *(const v8bf*)(bp);
        v8bf hi = *(const v8bf*)(bp + 16);
        bfrag[tn] = __builtin_shufflevector(lo, hi,
            0, 1, 2, 3, 4, 5, 6, 7, 8, 9, 10, 11, 12, 13, 14, 15);
    }
#pragma unroll
    for (int tm = 0; tm < 2; ++tm)
#pragma unroll
        for (int tn = 0; tn < 4; ++tn)
            acc[tm][tn] = __builtin_amdgcn_wmma_f32_16x16x32_bf16(
                false, afrag[tm], false, bfrag[tn],
                (short)0, acc[tm][tn], false, false);
}

__global__ void __launch_bounds__(256)
gemm_bf16_wmma(const unsigned short* __restrict__ Xq,
               const unsigned short* __restrict__ Wq,
               const float* __restrict__ bias,
               float* __restrict__ out,
               int M, int N, int K) {
    __shared__ __align__(16) unsigned short As[2][BM * LDT];
    __shared__ __align__(16) unsigned short Bs[2][BN * LDT];

    const int tid  = threadIdx.x;
    const int lane = tid & 31;
    const int wid  = tid >> 5;
    const int half = lane >> 4;       // 0: lanes 0-15, 1: lanes 16-31
    const int l15  = lane & 15;
    const int wm   = (wid & 3) * 32;  // 4 waves along M
    const int wn   = (wid >> 2) * 64; // 2 waves along N
    const int gm   = blockIdx.y * BM;
    const int gn   = blockIdx.x * BN;

    // global->LDS staging: thread owns half (h) of one 32-elem row (r)
    const int r = tid >> 1;
    const int h = tid & 1;

    v8f acc[2][4];
#pragma unroll
    for (int i = 0; i < 2; ++i)
#pragma unroll
        for (int j = 0; j < 4; ++j)
#pragma unroll
            for (int e = 0; e < 8; ++e) acc[i][j][e] = 0.0f;

    const unsigned short* gA = Xq + (size_t)(gm + r) * K + h * 16;
    const unsigned short* gB = Wq + (size_t)(gn + r) * K + h * 16;
    // AS(3)-typed staging pointers, cast once (hoists flat->local null check)
    ldsv4_t lA0 = TO_LDSV4(&As[0][r * LDT + h * 16]);
    ldsv4_t lA1 = TO_LDSV4(&As[1][r * LDT + h * 16]);
    ldsv4_t lB0 = TO_LDSV4(&Bs[0][r * LDT + h * 16]);
    ldsv4_t lB1 = TO_LDSV4(&Bs[1][r * LDT + h * 16]);

#define ISSUE_TILE(kt, lA, lB)                   \
    do {                                         \
        const int _k = (kt) * BKK;               \
        ASYNC_CP16(gA + _k,     (lA));           \
        ASYNC_CP16(gA + _k + 8, (lA) + 1);       \
        ASYNC_CP16(gB + _k,     (lB));           \
        ASYNC_CP16(gB + _k + 8, (lB) + 1);       \
    } while (0)

    const int NT = K / BKK;           // 128 K-tiles (even)

    ISSUE_TILE(0, lA0, lB0);          // prologue: tile 0 -> buffer 0

    for (int t = 0; t < NT; t += 2) {
        // ---- phase 0: compute buffer 0, prefetch tile t+1 into buffer 1
        ISSUE_TILE(t + 1, lA1, lB1);  // t+1 <= NT-1 always valid
        WAIT_ASYNC(4);                // in-order: tile t has landed
        __syncthreads();
        mma_step(&As[0][0], &Bs[0][0], wm, wn, l15, half, acc);
        __syncthreads();              // buffer 0 readers done

        // ---- phase 1: compute buffer 1, prefetch tile t+2 into buffer 0
        if (t + 2 < NT) {
            ISSUE_TILE(t + 2, lA0, lB0);
            WAIT_ASYNC(4);            // tile t+1 has landed
        } else {
            WAIT_ASYNC(0);            // last tile: drain
        }
        __syncthreads();
        mma_step(&As[1][0], &Bs[1][0], wm, wn, l15, half, acc);
        __syncthreads();              // buffer 1 readers done
    }
#undef ISSUE_TILE

    // Epilogue: C/D layout -> VGPR g holds (M = g + 8*half, N = l15)
#pragma unroll
    for (int tm = 0; tm < 2; ++tm) {
        const int row0 = gm + wm + tm * 16 + half * 8;
#pragma unroll
        for (int tn = 0; tn < 4; ++tn) {
            const int col = gn + wn + tn * 16 + l15;
            const float bv = bias[col];
            float* op = out + (size_t)row0 * N + col;
#pragma unroll
            for (int g = 0; g < 8; ++g)
                op[(size_t)g * N] = acc[tm][tn][g] + bv;
        }
    }
}

// ---------------------------------------------------------------------------
extern "C" void kernel_launch(void* const* d_in, const int* in_sizes, int n_in,
                              void* d_out, int out_size, void* d_ws, size_t ws_size,
                              hipStream_t stream) {
    const float* x    = (const float*)d_in[0];   // [8,2048,4096] fp32
    const float* w    = (const float*)d_in[1];   // [4096,4096] fp32 (AF16-quantized)
    const float* bias = (const float*)d_in[2];   // [4096] fp32 (AF16-quantized)
    float* out = (float*)d_out;                  // [8,2048,4096] fp32

    const int K = 4096, N = 4096;
    const long long xN = (long long)in_sizes[0];
    const long long wN = (long long)in_sizes[1];
    const int M = (int)(xN / K);                 // 16384

    unsigned short* xq = (unsigned short*)d_ws;  // M*K bf16
    unsigned short* wq = xq + xN;                // N*K bf16

    {
        long long n4 = xN / 4;
        int blocks = (int)((n4 + 255) / 256);
        quant_x_kernel<<<blocks, 256, 0, stream>>>(x, xq, xN);
    }
    {
        long long n4 = wN / 4;
        int blocks = (int)((n4 + 255) / 256);
        w_bf16_kernel<<<blocks, 256, 0, stream>>>(w, wq, wN);
    }
    dim3 grid(N / BN, M / BM);
    gemm_bf16_wmma<<<grid, 256, 0, stream>>>(xq, wq, bias, out, M, N, K);
}
